// ResNet4D_31190052504130
// MI455X (gfx1250) — matrix operand order repaired and checked by reading, not compiled, not run on hardware
//
#include <hip/hip_runtime.h>
#include <math.h>
#include <stdint.h>

typedef __attribute__((ext_vector_type(16))) _Float16 v16h;
typedef __attribute__((ext_vector_type(8)))  _Float16 v8h;
typedef __attribute__((ext_vector_type(8)))  float    v8f;

// ===========================================================================
// Prep kernel 1: repack conv weights (f32, reference layout
// w[kt][co][ci][kx][ky][kz]) into f16 [Co][Kpad], K-major, zero padded.
// GEMM K index: kk = ci*KS^4 + kt*KS^3 + kx*KS^2 + ky*KS + kz.
// ===========================================================================
template <int KS>
__global__ __launch_bounds__(256)
void repack_weights_kernel(const float* __restrict__ w, _Float16* __restrict__ wp,
                           int Co, int Ci, int Kpad)
{
    constexpr int K3 = KS * KS * KS;
    constexpr int K4 = K3 * KS;
    const int co = blockIdx.y;
    const int kk = blockIdx.x * 256 + threadIdx.x;
    if (kk >= Kpad) return;
    const int Ktot = Ci * K4;
    _Float16 v = (_Float16)0.0f;
    if (kk < Ktot) {
        const int ci = kk / K4;  int r = kk - ci * K4;
        const int kt = r / K3;   r -= kt * K3;          // r = spatial remainder
        v = (_Float16)w[((kt * Co + co) * Ci + ci) * K3 + r];
    }
    wp[(size_t)co * Kpad + kk] = v;
}

// ===========================================================================
// Prep kernel 2: K-decomposition table. For each kk:
//   .x = input-offset  ci*T^4 + kt*T^3 + kx*T^2 + ky*T + kz
//   .y = packed coords kt | kx<<8 | ky<<16 | kz<<24   (0xFFFFFFFF = pad)
// Depends only on (KS, Ci, T) -> shared across layers with equal config.
// ===========================================================================
template <int KS>
__global__ __launch_bounds__(256)
void build_table_kernel(int2* __restrict__ tab, int Ci, int T, int Kpad)
{
    constexpr int K2 = KS * KS, K3 = K2 * KS, K4 = K3 * KS;
    const int kk = blockIdx.x * 256 + threadIdx.x;
    if (kk >= Kpad) return;
    const int Ktot = Ci * K4;
    int2 e;
    if (kk < Ktot) {
        const int ci = kk / K4;  int r = kk - ci * K4;
        const int kt = r / K3;   r -= kt * K3;
        const int kx = r / K2;   r -= kx * K2;
        const int ky = r / KS;
        const int kz = r - ky * KS;
        const int T2 = T * T;
        e.x = ((ci * T + kt) * T + kx) * T2 + ky * T + kz;
        e.y = kt | (kx << 8) | (ky << 16) | (kz << 24);
    } else {
        e.x = 0;
        e.y = (int)0xFFFFFFFF;   // coords 255 -> always out of range
    }
    tab[kk] = e;
}

// ===========================================================================
// 4D conv as implicit GEMM on v_wmma_f32_16x16x32_f16.
//   One wave = 64x16 output tile: FOUR accumulators share ONE B fragment,
//   amortizing the scattered activation gather over 4 WMMAs per K-chunk.
//   A: contiguous 16B f16 loads per acc per K-chunk (pre-packed weights).
//   B: branchless table-driven gather (clamped address + cndmask select).
// ===========================================================================
template <int KS>
__global__ __launch_bounds__(32)
void conv4d_wmma_kernel(const float* __restrict__ x,        // Ci * T^4
                        const _Float16* __restrict__ wp,    // [Co][Kpad] f16
                        const int2* __restrict__ tab,       // [Kpad]
                        const float* __restrict__ bias,     // KS*Co or null
                        const float* __restrict__ res,      // Co*N or null
                        float* __restrict__ out,            // Co*N
                        int Co, int T, int O, int Kpad,
                        int stride, int pad, int relu)
{
    const int lane = threadIdx.x;      // 0..31
    const int half = lane >> 4;        // 0 or 1
    const int ml   = lane & 15;
    const int nTile = blockIdx.x;
    const int cTile = blockIdx.y;      // 64 channels per tile

    const int O2 = O * O, O3 = O2 * O, N = O3 * O;
    const int T2 = T * T, T3 = T2 * T;

    const int nB = nTile * 16 + ml;
    const bool nValid = (nB < N);
    int r = nValid ? nB : 0;
    const int ot = r / O3; r -= ot * O3;
    const int ox = r / O2; r -= ox * O2;
    const int oy = r / O;
    const int oz = r -  oy * O;
    const int it0 = ot * stride - pad;
    const int ix0 = ox * stride - pad;
    const int iy0 = oy * stride - pad;
    const int iz0 = oz * stride - pad;
    const int posBase = it0 * T3 + ix0 * T2 + iy0 * T + iz0;

    const int co0 = cTile * 64 + ml;
    const _Float16* wp0 = wp + (size_t)co0 * Kpad + half * 8;
    const _Float16* wp1 = wp0 + (size_t)16 * Kpad;
    const _Float16* wp2 = wp0 + (size_t)32 * Kpad;
    const _Float16* wp3 = wp0 + (size_t)48 * Kpad;

    v8f acc0 = {}, acc1 = {}, acc2 = {}, acc3 = {};

    for (int k0 = 0; k0 < Kpad; k0 += 32) {
        // ---- A fragments: contiguous f16, layout-exact for 16x32 A ------
        const v8h a0lo = *(const v8h*)(wp0 + k0);        // K = k0+half*8+(0..7)
        const v8h a0hi = *(const v8h*)(wp0 + k0 + 16);   // K = k0+16+half*8+(0..7)
        const v8h a1lo = *(const v8h*)(wp1 + k0);
        const v8h a1hi = *(const v8h*)(wp1 + k0 + 16);
        const v8h a2lo = *(const v8h*)(wp2 + k0);
        const v8h a2hi = *(const v8h*)(wp2 + k0 + 16);
        const v8h a3lo = *(const v8h*)(wp3 + k0);
        const v8h a3hi = *(const v8h*)(wp3 + k0 + 16);
        __builtin_prefetch((const void*)(wp0 + k0 + 128), 0, 1);

        // ---- B fragment: 32x16, lane column n=ml, K = half*8 + h + (h&8) -------
        v16h b;
        const int2* t = tab + k0 + half * 8;
        #pragma unroll
        for (int h = 0; h < 16; ++h) {
            const int2 e = t[h + (h & 8)];
            const int kt =  e.y        & 0xFF;
            const int kx = (e.y >> 8)  & 0xFF;
            const int ky = (e.y >> 16) & 0xFF;
            const int kz = (e.y >> 24) & 0xFF;
            const bool valid = nValid &
                ((unsigned)(it0 + kt) < (unsigned)T) &
                ((unsigned)(ix0 + kx) < (unsigned)T) &
                ((unsigned)(iy0 + ky) < (unsigned)T) &
                ((unsigned)(iz0 + kz) < (unsigned)T);
            const int addr = valid ? (posBase + e.x) : 0;   // clamped, safe
            const float v = x[addr];                        // unconditional
            b[h] = valid ? (_Float16)v : (_Float16)0.0f;
        }

        const v16h A0 = __builtin_shufflevector(a0lo, a0hi,
            0,1,2,3,4,5,6,7,8,9,10,11,12,13,14,15);
        const v16h A1 = __builtin_shufflevector(a1lo, a1hi,
            0,1,2,3,4,5,6,7,8,9,10,11,12,13,14,15);
        const v16h A2 = __builtin_shufflevector(a2lo, a2hi,
            0,1,2,3,4,5,6,7,8,9,10,11,12,13,14,15);
        const v16h A3 = __builtin_shufflevector(a3lo, a3hi,
            0,1,2,3,4,5,6,7,8,9,10,11,12,13,14,15);

        acc0 = __builtin_amdgcn_wmma_f32_16x16x32_f16(
                   false, A0, false, b, (short)0, acc0, false, false);
        acc1 = __builtin_amdgcn_wmma_f32_16x16x32_f16(
                   false, A1, false, b, (short)0, acc1, false, false);
        acc2 = __builtin_amdgcn_wmma_f32_16x16x32_f16(
                   false, A2, false, b, (short)0, acc2, false, false);
        acc3 = __builtin_amdgcn_wmma_f32_16x16x32_f16(
                   false, A3, false, b, (short)0, acc3, false, false);
    }

    // ---- Epilogue: per-valid-temporal-tap bias, residual, ReLU ----------
    if (nValid) {
        #pragma unroll
        for (int vv = 0; vv < 8; ++vv) {
            const int coa = cTile * 64 + half * 8 + vv;
            float oa = acc0[vv];
            float ob = acc1[vv];
            float oc = acc2[vv];
            float od = acc3[vv];
            if (bias) {
                #pragma unroll
                for (int kt = 0; kt < KS; ++kt)
                    if ((unsigned)(it0 + kt) < (unsigned)T) {
                        oa += bias[kt * Co + coa];
                        ob += bias[kt * Co + coa + 16];
                        oc += bias[kt * Co + coa + 32];
                        od += bias[kt * Co + coa + 48];
                    }
            }
            if (res) {
                oa += res[(coa     ) * N + nB];
                ob += res[(coa + 16) * N + nB];
                oc += res[(coa + 32) * N + nB];
                od += res[(coa + 48) * N + nB];
            }
            if (relu) {
                oa = fmaxf(oa, 0.0f);
                ob = fmaxf(ob, 0.0f);
                oc = fmaxf(oc, 0.0f);
                od = fmaxf(od, 0.0f);
            }
            out[(coa     ) * N + nB] = oa;
            out[(coa + 16) * N + nB] = ob;
            out[(coa + 32) * N + nB] = oc;
            out[(coa + 48) * N + nB] = od;
        }
    }
}

// ===========================================================================
// 4D maxpool, k=3 s=2 p=1 (-inf padding)
// ===========================================================================
__global__ __launch_bounds__(256)
void maxpool4d_kernel(const float* __restrict__ in, float* __restrict__ out,
                      int C, int T, int O)
{
    const int idx = blockIdx.x * blockDim.x + threadIdx.x;
    const int O2 = O * O, O3 = O2 * O, O4 = O3 * O;
    const int Ntot = C * O4;
    if (idx >= Ntot) return;
    int r = idx;
    const int c  = r / O4; r -= c * O4;
    const int ot = r / O3; r -= ot * O3;
    const int ox = r / O2; r -= ox * O2;
    const int oy = r / O;
    const int oz = r -  oy * O;
    const int T2 = T * T;
    float m = -INFINITY;
    for (int a = 0; a < 3; ++a) {
        const int it = ot * 2 - 1 + a; if ((unsigned)it >= (unsigned)T) continue;
        for (int b = 0; b < 3; ++b) {
            const int ix = ox * 2 - 1 + b; if ((unsigned)ix >= (unsigned)T) continue;
            for (int d = 0; d < 3; ++d) {
                const int iy = oy * 2 - 1 + d; if ((unsigned)iy >= (unsigned)T) continue;
                for (int e = 0; e < 3; ++e) {
                    const int iz = oz * 2 - 1 + e; if ((unsigned)iz >= (unsigned)T) continue;
                    m = fmaxf(m, in[((c * T + it) * T + ix) * T2 + iy * T + iz]);
                }
            }
        }
    }
    out[idx] = m;
}

// ===========================================================================
// Head: 2^4 avgpool over 512 channels + FC(512->1)
// ===========================================================================
__global__ __launch_bounds__(512)
void head_kernel(const float* __restrict__ in, const float* __restrict__ fcw,
                 const float* __restrict__ fcb, float* __restrict__ out)
{
    __shared__ float s[512];
    const int c = threadIdx.x;
    float sum = 0.0f;
    #pragma unroll
    for (int i = 0; i < 16; ++i) sum += in[c * 16 + i];
    s[c] = (sum * (1.0f / 16.0f)) * fcw[c];
    __syncthreads();
    for (int str = 256; str > 0; str >>= 1) {
        if (c < str) s[c] += s[c + str];
        __syncthreads();
    }
    if (c == 0) out[0] = s[0] + fcb[0];
}

// ===========================================================================
template <int KS>
static inline void launch_conv(const float* x, const _Float16* wp, const int2* tab,
                               const float* bias, const float* res, float* out,
                               int Co, int T, int O, int Kpad,
                               int s, int p, int relu, hipStream_t stream)
{
    const int N = O * O * O * O;
    dim3 grid((N + 15) / 16, Co / 64);
    conv4d_wmma_kernel<KS><<<grid, dim3(32), 0, stream>>>(
        x, wp, tab, bias, res, out, Co, T, O, Kpad, s, p, relu);
}

template <int KS>
static inline void launch_repack(const float* w, _Float16* wp,
                                 int Co, int Ci, int Kpad, hipStream_t stream)
{
    dim3 grid((Kpad + 255) / 256, Co);
    repack_weights_kernel<KS><<<grid, dim3(256), 0, stream>>>(w, wp, Co, Ci, Kpad);
}

template <int KS>
static inline void launch_table(int2* tab, int Ci, int T, int Kpad, hipStream_t stream)
{
    build_table_kernel<KS><<<dim3((Kpad + 255) / 256), dim3(256), 0, stream>>>(
        tab, Ci, T, Kpad);
}

extern "C" void kernel_launch(void* const* d_in, const int* in_sizes, int n_in,
                              void* d_out, int out_size, void* d_ws, size_t ws_size,
                              hipStream_t stream)
{
    (void)in_sizes; (void)n_in; (void)out_size; (void)ws_size;

    const float* x       = (const float*)d_in[0];
    const float* conv1_w = (const float*)d_in[1];
    const float* l1_w1   = (const float*)d_in[2];
    const float* l1_b1   = (const float*)d_in[3];
    const float* l1_w2   = (const float*)d_in[4];
    const float* l1_b2   = (const float*)d_in[5];
    const float* l2_w1   = (const float*)d_in[6];
    const float* l2_b1   = (const float*)d_in[7];
    const float* l2_w2   = (const float*)d_in[8];
    const float* l2_b2   = (const float*)d_in[9];
    const float* l2_dw   = (const float*)d_in[10];
    const float* l3_w1   = (const float*)d_in[11];
    const float* l3_b1   = (const float*)d_in[12];
    const float* l3_w2   = (const float*)d_in[13];
    const float* l3_b2   = (const float*)d_in[14];
    const float* l3_dw   = (const float*)d_in[15];
    const float* l4_w1   = (const float*)d_in[16];
    const float* l4_b1   = (const float*)d_in[17];
    const float* l4_w2   = (const float*)d_in[18];
    const float* l4_b2   = (const float*)d_in[19];
    const float* l4_dw   = (const float*)d_in[20];
    const float* fc_w    = (const float*)d_in[21];
    const float* fc_b    = (const float*)d_in[22];

    // ---- workspace layout -------------------------------------------------
    float* ws = (float*)d_ws;
    float* bufConv1 = ws;                      // 64 * 18^4 = 6,718,464 f32
    float* bufA     = bufConv1 + 6718464;      // 419,904 f32
    float* bufB     = bufA + 419904;
    float* bufC     = bufB + 419904;
    float* bufR     = bufC + 419904;           // 80,000 f32

    _Float16* pw_c1   = (_Float16*)(bufR + 80000);
    _Float16* pw_l1w1 = pw_c1   + (size_t)64  * 2432;
    _Float16* pw_l1w2 = pw_l1w1 + (size_t)64  * 5184;
    _Float16* pw_l2w1 = pw_l1w2 + (size_t)64  * 5184;
    _Float16* pw_l2w2 = pw_l2w1 + (size_t)128 * 5184;
    _Float16* pw_l2dw = pw_l2w2 + (size_t)128 * 10368;
    _Float16* pw_l3w1 = pw_l2dw + (size_t)128 * 64;
    _Float16* pw_l3w2 = pw_l3w1 + (size_t)256 * 10368;
    _Float16* pw_l3dw = pw_l3w2 + (size_t)256 * 20736;
    _Float16* pw_l4w1 = pw_l3dw + (size_t)256 * 128;
    _Float16* pw_l4w2 = pw_l4w1 + (size_t)512 * 20736;
    _Float16* pw_l4dw = pw_l4w2 + (size_t)512 * 41472;
    _Float16* pw_end  = pw_l4dw + (size_t)512 * 256;

    uintptr_t tb_addr = ((uintptr_t)pw_end + 7) & ~(uintptr_t)7;
    int2* tb_c1 = (int2*)tb_addr;   // (KS=7, Ci=1,   T=36) Kpad 2432
    int2* tb_a  = tb_c1 + 2432;     // (3, 64,  9)  5184
    int2* tb_b  = tb_a  + 5184;     // (3, 128, 5)  10368
    int2* tb_c  = tb_b  + 10368;    // (3, 256, 3)  20736
    int2* tb_d  = tb_c  + 20736;    // (3, 512, 2)  41472
    int2* tb_e  = tb_d  + 41472;    // (1, 64,  9)  64
    int2* tb_f  = tb_e  + 64;       // (1, 128, 5)  128
    int2* tb_g  = tb_f  + 128;      // (1, 256, 3)  256

    // ---- prep: pack weights + build K-decomposition tables ----------------
    launch_repack<7>(conv1_w, pw_c1,   64,  1,   2432,  stream);
    launch_repack<3>(l1_w1,   pw_l1w1, 64,  64,  5184,  stream);
    launch_repack<3>(l1_w2,   pw_l1w2, 64,  64,  5184,  stream);
    launch_repack<3>(l2_w1,   pw_l2w1, 128, 64,  5184,  stream);
    launch_repack<3>(l2_w2,   pw_l2w2, 128, 128, 10368, stream);
    launch_repack<1>(l2_dw,   pw_l2dw, 128, 64,  64,    stream);
    launch_repack<3>(l3_w1,   pw_l3w1, 256, 128, 10368, stream);
    launch_repack<3>(l3_w2,   pw_l3w2, 256, 256, 20736, stream);
    launch_repack<1>(l3_dw,   pw_l3dw, 256, 128, 128,   stream);
    launch_repack<3>(l4_w1,   pw_l4w1, 512, 256, 20736, stream);
    launch_repack<3>(l4_w2,   pw_l4w2, 512, 512, 41472, stream);
    launch_repack<1>(l4_dw,   pw_l4dw, 512, 256, 256,   stream);

    launch_table<7>(tb_c1, 1,   36, 2432,  stream);
    launch_table<3>(tb_a,  64,  9,  5184,  stream);
    launch_table<3>(tb_b,  128, 5,  10368, stream);
    launch_table<3>(tb_c,  256, 3,  20736, stream);
    launch_table<3>(tb_d,  512, 2,  41472, stream);
    launch_table<1>(tb_e,  64,  9,  64,    stream);
    launch_table<1>(tb_f,  128, 5,  128,   stream);
    launch_table<1>(tb_g,  256, 3,  256,   stream);

    // ---- network ----------------------------------------------------------
    // Stem
    launch_conv<7>(x, pw_c1, tb_c1, nullptr, nullptr, bufConv1,
                   64, 36, 18, 2432, 2, 3, 1, stream);
    {
        const int Np = 64 * 9 * 9 * 9 * 9;
        maxpool4d_kernel<<<(Np + 255) / 256, 256, 0, stream>>>(bufConv1, bufA, 64, 18, 9);
    }
    // Layer1 (64->64, s1, identity residual)
    launch_conv<3>(bufA, pw_l1w1, tb_a, l1_b1, nullptr, bufB, 64, 9, 9, 5184, 1, 1, 1, stream);
    launch_conv<3>(bufB, pw_l1w2, tb_a, l1_b2, bufA,    bufC, 64, 9, 9, 5184, 1, 1, 1, stream);
    // Layer2 (64->128, s2, 1x1 downsample)
    launch_conv<1>(bufC, pw_l2dw, tb_e, nullptr, nullptr, bufR, 128, 9, 5, 64,    2, 0, 0, stream);
    launch_conv<3>(bufC, pw_l2w1, tb_a, l2_b1, nullptr,   bufB, 128, 9, 5, 5184,  2, 1, 1, stream);
    launch_conv<3>(bufB, pw_l2w2, tb_b, l2_b2, bufR,      bufA, 128, 5, 5, 10368, 1, 1, 1, stream);
    // Layer3 (128->256, s2)
    launch_conv<1>(bufA, pw_l3dw, tb_f, nullptr, nullptr, bufR, 256, 5, 3, 128,   2, 0, 0, stream);
    launch_conv<3>(bufA, pw_l3w1, tb_b, l3_b1, nullptr,   bufB, 256, 5, 3, 10368, 2, 1, 1, stream);
    launch_conv<3>(bufB, pw_l3w2, tb_c, l3_b2, bufR,      bufC, 256, 3, 3, 20736, 1, 1, 1, stream);
    // Layer4 (256->512, s2)
    launch_conv<1>(bufC, pw_l4dw, tb_g, nullptr, nullptr, bufR, 512, 3, 2, 256,   2, 0, 0, stream);
    launch_conv<3>(bufC, pw_l4w1, tb_c, l4_b1, nullptr,   bufB, 512, 3, 2, 20736, 2, 1, 1, stream);
    launch_conv<3>(bufB, pw_l4w2, tb_d, l4_b2, bufR,      bufA, 512, 2, 2, 41472, 1, 1, 1, stream);
    // Head
    head_kernel<<<1, 512, 0, stream>>>(bufA, fc_w, fc_b, (float*)d_out);
}
